// Attend_73134703117052
// MI455X (gfx1250) — matrix operand order repaired and checked by reading, not compiled, not run on hardware
//
#include <hip/hip_runtime.h>

typedef __attribute__((ext_vector_type(16))) _Float16 v16h;
typedef __attribute__((ext_vector_type(8)))  _Float16 v8h;
typedef __attribute__((ext_vector_type(8)))  float    v8f;
typedef __attribute__((ext_vector_type(4)))  unsigned int u32x4;
typedef __attribute__((ext_vector_type(8)))  int      i32x8;
typedef __attribute__((ext_vector_type(4)))  int      i32x4;

constexpr int kB   = 2;
constexpr int kH   = 32;
constexpr int kKVH = 8;
constexpr int kN   = 2048;
constexpr int kD   = 128;
constexpr int kQT  = 128;   // q rows per workgroup (8 waves x 16)
constexpr int kKT  = 64;    // kv columns per tile
constexpr int kKSTR = 136;  // LDS row stride (halves) for K tile   (256B row + 16B pad)
constexpr int kVSTR = 72;   // LDS row stride (halves) for V^T tile (128B row + 16B pad)
constexpr float kScale = 0.088388347648318447f;                    // 1/sqrt(128)
constexpr float kScaleLog2e = kScale * 1.4426950408889634f;        // folded into Q
constexpr float kNegBig = -3.0e38f;

constexpr size_t kKVElems = (size_t)kB * kKVH * kN * kD;   // elems per K (or V) tensor

#define USE_TDM __has_builtin(__builtin_amdgcn_tensor_load_to_lds)

__device__ __forceinline__ v8f wmma_f16(v16h a, v16h b, v8f c) {
  return __builtin_amdgcn_wmma_f32_16x16x32_f16(false, a, false, b, (short)0, c,
                                                false, false);
}

// A-fragment (16x32 f16), contiguous k: lanes 0-15 k 0..15, lanes 16-31 k 16..31.
__device__ __forceinline__ v16h load_a_frag(const _Float16* row, int hi) {
  union { v16h v; v8h h[2]; } u;
  u.h[0] = *(const v8h*)(row + hi * 16);
  u.h[1] = *(const v8h*)(row + hi * 16 + 8);
  return u.v;
}

// A-fragment for the PV matmul in table order (matches P^T fed from C layout:
// lanes 0-15 kv 0..7,16..23; lanes 16-31 kv 8..15,24..31).
__device__ __forceinline__ v16h load_av_perm(const _Float16* row, int hi) {
  union { v16h v; v8h h[2]; } u;
  const _Float16* p = row + hi * 8;
  u.h[0] = *(const v8h*)(p);
  u.h[1] = *(const v8h*)(p + 16);
  return u.v;
}

#if USE_TDM
// Issue a 2D TDM tile load: tensor dims == tile dims (always fully in-bounds),
// data_size = 2B, LDS row padding to reproduce padded LDS strides.
__device__ __forceinline__ void tdm_load_2d(unsigned lds_off, const void* gaddr,
                                            unsigned dim0, unsigned dim1,
                                            unsigned stride0,
                                            unsigned pad_interval,
                                            unsigned pad_amount) {
  const unsigned long long ga = (unsigned long long)gaddr;
  u32x4 g0;
  g0[0] = 1u;                                   // count=1, user descriptor
  g0[1] = lds_off;                              // lds_addr (bytes)
  g0[2] = (unsigned)(ga & 0xFFFFFFFFu);         // global_addr[31:0]
  g0[3] = (unsigned)((ga >> 32) & 0x01FFFFFFu)  // global_addr[56:32]
          | (2u << 30);                         // type = 2 ("image")
  i32x8 g1;
  g1[0] = (int)((1u << 16)                      // data_size = 2 bytes
                | (1u << 20)                    // pad_enable
                | (pad_interval << 22) | (pad_amount << 25));
  g1[1] = (int)((dim0 & 0xFFFFu) << 16);                        // tensor_dim0 lo
  g1[2] = (int)(((dim0 >> 16) & 0xFFFFu) | ((dim1 & 0xFFFFu) << 16));
  g1[3] = (int)(((dim1 >> 16) & 0xFFFFu) | ((dim0 & 0xFFFFu) << 16)); // tile_dim0=dim0
  g1[4] = (int)(dim1 & 0xFFFFu);                                // tile_dim1=dim1
  g1[5] = (int)stride0;                                         // tensor_dim0_stride
  g1[6] = 0;
  g1[7] = 0;
  i32x4 z4 = (i32x4)0;
  i32x8 z8 = (i32x8)0;
  __builtin_amdgcn_tensor_load_to_lds(g0, g1, z4, z4, z8, 0);
}
#endif

// ---------------- Pre-pass kernels: f32 -> f16 (and V -> V^T) in d_ws --------

__global__ __launch_bounds__(256, 1)
void prep_k_kernel(const float* __restrict__ kg, _Float16* __restrict__ kh) {
  const size_t i = ((size_t)blockIdx.x * 256 + threadIdx.x) * 8;
  float4 f0 = *(const float4*)(kg + i);
  float4 f1 = *(const float4*)(kg + i + 4);
  v8h o;
  o[0] = (_Float16)f0.x; o[1] = (_Float16)f0.y;
  o[2] = (_Float16)f0.z; o[3] = (_Float16)f0.w;
  o[4] = (_Float16)f1.x; o[5] = (_Float16)f1.y;
  o[6] = (_Float16)f1.z; o[7] = (_Float16)f1.w;
  *(v8h*)(kh + i) = o;
}

__global__ __launch_bounds__(256, 1)
void prep_v_kernel(const float* __restrict__ vg, _Float16* __restrict__ vth) {
  __shared__ __align__(16) _Float16 tile[kD * kVSTR];   // d x 64kv (padded)
  const int tid  = threadIdx.x;
  const int slab = blockIdx.x >> 5;            // (b*KVH + hkv)
  const int kvc  = (blockIdx.x & 31) * 64;     // kv chunk
  const float* vp = vg + (size_t)slab * kN * kD;
  {
    const int d0  = (tid & 31) * 4;
    const int kvb = (tid >> 5) * 8;
    v8h col[4];
#pragma unroll
    for (int i = 0; i < 8; ++i) {
      float4 f = *(const float4*)(vp + (size_t)(kvc + kvb + i) * kD + d0);
      col[0][i] = (_Float16)f.x;
      col[1][i] = (_Float16)f.y;
      col[2][i] = (_Float16)f.z;
      col[3][i] = (_Float16)f.w;
    }
#pragma unroll
    for (int j = 0; j < 4; ++j)
      *(v8h*)(&tile[(d0 + j) * kVSTR + kvb]) = col[j];
  }
  __syncthreads();
  {
    const int d  = tid >> 1;
    const int h0 = (tid & 1) * 32;
    _Float16* dst = vth + (size_t)slab * kD * kN + (size_t)d * kN + kvc + h0;
    const _Float16* src = &tile[d * kVSTR + h0];
#pragma unroll
    for (int j = 0; j < 4; ++j)
      *(v8h*)(dst + 8 * j) = *(const v8h*)(src + 8 * j);
  }
}

// ---------------- Main flash-attention kernel --------------------------------

__global__ __launch_bounds__(256, 1)
void fattn_kernel(const float* __restrict__ qg, const _Float16* __restrict__ kh,
                  const _Float16* __restrict__ vth, float* __restrict__ og) {
  __shared__ __align__(16) _Float16 ksh[2][kKT * kKSTR];   // K tile (double buffer)
  __shared__ __align__(16) _Float16 vtsh[2][kD * kVSTR];   // V^T tile (double buffer)

  const int tid  = threadIdx.x;
  const int lane = tid & 31;
  const int wave = tid >> 5;
  const int hi   = lane >> 4;
  const int ln16 = lane & 15;

  const int blk = blockIdx.x;
  const int qb  = blk % (kN / kQT);
  const int bh  = blk / (kN / kQT);
  const int h   = bh % kH;
  const int b   = bh / kH;
  const int hkv = h / (kH / kKVH);

  const float*    qp = qg + (size_t)bh * kN * kD;
  const _Float16* ks = kh + (size_t)(b * kKVH + hkv) * kN * kD;   // f16 K slab
  const _Float16* vs = vth + (size_t)(b * kKVH + hkv) * kD * kN;  // f16 V^T slab
  float*          op = og + (size_t)bh * kN * kD;

  const int q0   = qb * kQT + wave * 16;
  const int qrow = q0 + ln16;

  // ---- Q^T B-fragments, scaled by SCALE*log2(e): 4 d-slices of 32 ----
  v16h qf[4];
  {
    const float* qr = qp + (size_t)qrow * kD + hi * 16;
#pragma unroll
    for (int s = 0; s < 4; ++s) {
      const float4* p4 = (const float4*)(qr + s * 32);
#pragma unroll
      for (int c = 0; c < 4; ++c) {
        float4 f = p4[c];
        qf[s][4 * c + 0] = (_Float16)(f.x * kScaleLog2e);
        qf[s][4 * c + 1] = (_Float16)(f.y * kScaleLog2e);
        qf[s][4 * c + 2] = (_Float16)(f.z * kScaleLog2e);
        qf[s][4 * c + 3] = (_Float16)(f.w * kScaleLog2e);
      }
    }
  }

  v8f acc[8];
#pragma unroll
  for (int t = 0; t < 8; ++t) acc[t] = (v8f)0.0f;
  float mrow = kNegBig;
  float lrow = 0.0f;

  const int nkv = (qb + 1) * kQT;

  // Tile fill: TDM (wave 0 issues async DMA) or cooperative f16 copy fallback.
  auto fill_tile = [&](int buf, int kv0) {
#if USE_TDM
    if (wave == 0) {
      tdm_load_2d((unsigned)(size_t)&ksh[buf][0], ks + (size_t)kv0 * kD,
                  /*dim0=*/kD, /*dim1=*/kKT, /*stride0=*/kD,
                  /*pad_interval=*/5, /*pad_amount=*/3);   // 256B rows + 16B pad
      tdm_load_2d((unsigned)(size_t)&vtsh[buf][0], vs + kv0,
                  /*dim0=*/kKT, /*dim1=*/kD, /*stride0=*/kN,
                  /*pad_interval=*/4, /*pad_amount=*/3);   // 128B rows + 16B pad
    }
#else
    {
      const int row = tid >> 2;
      const int c0  = (tid & 3) * 32;
      const _Float16* src = ks + (size_t)(kv0 + row) * kD + c0;
#pragma unroll
      for (int j = 0; j < 4; ++j)
        *(v8h*)(&ksh[buf][row * kKSTR + c0 + 8 * j]) = *(const v8h*)(src + 8 * j);
    }
    {
      const int d  = tid >> 1;
      const int h0 = (tid & 1) * 32;
      const _Float16* src = vs + (size_t)d * kN + kv0 + h0;
#pragma unroll
      for (int j = 0; j < 4; ++j)
        *(v8h*)(&vtsh[buf][d * kVSTR + h0 + 8 * j]) = *(const v8h*)(src + 8 * j);
    }
#endif
  };

  fill_tile(0, 0);   // prologue

  int t = 0;
  for (int kv0 = 0; kv0 < nkv; kv0 += kKT, ++t) {
    const int buf  = t & 1;
    const bool more = (kv0 + kKT < nkv);
    if (more) fill_tile(buf ^ 1, kv0 + kKT);   // async-issue next tile
#if USE_TDM
    if (wave == 0) {
      if (more) __builtin_amdgcn_s_wait_tensorcnt(2);  // current tile done (in-order)
      else      __builtin_amdgcn_s_wait_tensorcnt(0);
    }
#endif
    __syncthreads();

    const _Float16* kbuf = &ksh[buf][0];
    const _Float16* vbuf = &vtsh[buf][0];

    if (kv0 <= q0 + 15) {   // wave-uniform skip of fully-masked tiles
      // ---- S^T = K_tile x Q^T : 4 kv-subtiles x 4 d-slices, pipelined ----
      v8f s[4];
#pragma unroll
      for (int st = 0; st < 4; ++st) s[st] = (v8f)0.0f;

      v16h kfrag = load_a_frag(kbuf + ln16 * kKSTR, hi);
#pragma unroll
      for (int i = 0; i < 16; ++i) {
        const int st = i & 3;
        const int sl = i >> 2;
        v16h nxt = kfrag;
        if (i < 15) {
          const int st2 = (i + 1) & 3;
          const int sl2 = (i + 1) >> 2;
          nxt = load_a_frag(kbuf + (st2 * 16 + ln16) * kKSTR + sl2 * 32, hi);
        }
        s[st] = wmma_f16(kfrag, qf[sl], s[st]);
        kfrag = nxt;
      }

      // ---- Causal mask (diagonal-straddling tiles only) ----
      if (kv0 + kKT - 1 > q0) {
#pragma unroll
        for (int st = 0; st < 4; ++st)
#pragma unroll
          for (int r = 0; r < 8; ++r) {
            const int kv = kv0 + st * 16 + r + 8 * hi;
            s[st][r] = (kv > qrow) ? kNegBig : s[st][r];
          }
      }

      // ---- Online softmax in exp2 domain ----
      float tmax = s[0][0];
#pragma unroll
      for (int st = 0; st < 4; ++st)
#pragma unroll
        for (int r = 0; r < 8; ++r) tmax = fmaxf(tmax, s[st][r]);
      tmax = fmaxf(tmax, __shfl_xor(tmax, 16, 32));
      const float mnew = fmaxf(mrow, tmax);

      float ssum = 0.0f;
#pragma unroll
      for (int st = 0; st < 4; ++st)
#pragma unroll
        for (int r = 0; r < 8; ++r) {
          const float e = __builtin_exp2f(s[st][r] - mnew);
          s[st][r] = e;
          ssum += e;
        }
      ssum += __shfl_xor(ssum, 16, 32);

      if (__ballot(tmax > mrow) != 0ull) {   // wave-uniform rescale gate
        const float alpha = __builtin_exp2f(mrow - mnew);
        lrow = lrow * alpha + ssum;
        mrow = mnew;
#pragma unroll
        for (int u = 0; u < 8; ++u)
#pragma unroll
          for (int r = 0; r < 8; ++r) acc[u][r] *= alpha;
      } else {
        lrow += ssum;
      }

      // ---- P^T -> B-fragments straight from C layout (K-permuted) ----
      v16h pb0, pb1;
#pragma unroll
      for (int r = 0; r < 8; ++r) {
        pb0[r]     = (_Float16)s[0][r];
        pb0[8 + r] = (_Float16)s[1][r];
        pb1[r]     = (_Float16)s[2][r];
        pb1[8 + r] = (_Float16)s[3][r];
      }

      // ---- out^T += V^T(perm) x P^T(perm) : 8 d-tiles, pipelined ----
      v16h a0 = load_av_perm(vbuf + ln16 * kVSTR, hi);
      v16h a1 = load_av_perm(vbuf + ln16 * kVSTR + 32, hi);
#pragma unroll
      for (int dt = 0; dt < 8; ++dt) {
        v16h n0 = a0, n1 = a1;
        if (dt < 7) {
          const _Float16* nr = vbuf + ((dt + 1) * 16 + ln16) * kVSTR;
          n0 = load_av_perm(nr, hi);
          n1 = load_av_perm(nr + 32, hi);
        }
        acc[dt] = wmma_f16(a0, pb0, acc[dt]);
        acc[dt] = wmma_f16(a1, pb1, acc[dt]);
        a0 = n0;
        a1 = n1;
      }
    }
    __syncthreads();
  }

  // ---- Finalize ----
  const float inv = 1.0f / lrow;
  float* orow = op + (size_t)qrow * kD + 8 * hi;
#pragma unroll
  for (int dt = 0; dt < 8; ++dt) {
    float4 o0, o1;
    o0.x = acc[dt][0] * inv; o0.y = acc[dt][1] * inv;
    o0.z = acc[dt][2] * inv; o0.w = acc[dt][3] * inv;
    o1.x = acc[dt][4] * inv; o1.y = acc[dt][5] * inv;
    o1.z = acc[dt][6] * inv; o1.w = acc[dt][7] * inv;
    *(float4*)(orow + dt * 16)     = o0;
    *(float4*)(orow + dt * 16 + 4) = o1;
  }
}

extern "C" void kernel_launch(void* const* d_in, const int* in_sizes, int n_in,
                              void* d_out, int out_size, void* d_ws, size_t ws_size,
                              hipStream_t stream) {
  const float* q = (const float*)d_in[0];
  const float* k = (const float*)d_in[1];
  const float* v = (const float*)d_in[2];
  float* out = (float*)d_out;
  (void)in_sizes; (void)n_in; (void)out_size; (void)ws_size;

  _Float16* kh  = (_Float16*)d_ws;           // f16 K, row-major
  _Float16* vth = kh + kKVElems;             // f16 V^T, [slab][d][n]

  prep_k_kernel<<<dim3((unsigned)(kKVElems / 2048)), dim3(256), 0, stream>>>(k, kh);
  prep_v_kernel<<<dim3(kB * kKVH * (kN / 64)), dim3(256), 0, stream>>>(v, vth);

  fattn_kernel<<<dim3(kB * kH * (kN / kQT)), dim3(256), 0, stream>>>(q, kh, vth, out);
}
